// GAT_1262720385650
// MI455X (gfx1250) — compile-verified, matrix operand order checked by reading, not run on hardware
//
#include <hip/hip_runtime.h>
#include <hip/hip_bf16.h>

typedef _Float16 v16h __attribute__((ext_vector_type(16)));
typedef _Float16 v8h  __attribute__((ext_vector_type(8)));
typedef float    v8f  __attribute__((ext_vector_type(8)));

#define NEG_SLOPE 0.2f

// ---------------------------------------------------------------- utilities
__global__ void k_f32_to_f16(const float* __restrict__ in, _Float16* __restrict__ out, long n) {
    long i = (long)blockIdx.x * blockDim.x + threadIdx.x;
    if (i < n) out[i] = (_Float16)in[i];
}

__global__ void k_fill(float* __restrict__ p, float v, long n) {
    long i = (long)blockIdx.x * blockDim.x + threadIdx.x;
    if (i < n) p[i] = v;
}

// ----------------------------------------------------------------- WMMA GEMM
// C[M,Nout] = A[M,K] @ B[Nout,K]^T ; A,B f16 row-major (B = weight matrix W).
// One wave computes a 16x64 strip: A fragment loaded once per K-step, reused
// across 4 v_wmma_f32_16x16x32_f16 with 4 different B fragments.

__device__ __forceinline__ v16h load_frag(const _Float16* __restrict__ p, int khalf) {
    // ISA 16-bit A/B 16x32 per-lane layout: halves at K = khalf+[0,8) and 16+khalf+[0,8)
    v8h lo = *(const v8h*)(p + khalf);
    v8h hi = *(const v8h*)(p + 16 + khalf);
    v16h f;
#pragma unroll
    for (int i = 0; i < 8; ++i) { f[i] = lo[i]; f[8 + i] = hi[i]; }
    return f;
}

__device__ __forceinline__ v8f wmma_f16(v16h a, v16h b, v8f c) {
    return __builtin_amdgcn_wmma_f32_16x16x32_f16(
        /*neg_a=*/false, a, /*neg_b=*/false, b,
        /*c_mod=*/(short)0, c, /*reuse_a=*/false, /*reuse_b=*/false);
}

__global__ void k_gemm_wmma(const _Float16* __restrict__ A, const _Float16* __restrict__ B,
                            float* __restrict__ C, int Nout, int K,
                            int tilesN4, int totalWaves) {
    int wave = blockIdx.x * (blockDim.x >> 5) + (threadIdx.x >> 5);
    if (wave >= totalWaves) return;            // wave-uniform exit; EXEC stays all-ones
    int lane  = threadIdx.x & 31;
    int r     = lane & 15;                     // row-in-tile (A) / col-in-tile (B)
    int hi    = lane >> 4;                     // lane group 0/1
    int khalf = hi << 3;                       // 0 or 8: K base for this lane group

    int mt  = wave / tilesN4;                  // M tile
    int nt4 = wave - mt * tilesN4;             // group of 4 N tiles

    const _Float16* arow = A + (long)(mt * 16 + r) * K;
    const _Float16* brow = B + (long)(nt4 * 64 + r) * K;  // + j*16*K for tile j

    v8f acc0 = {}, acc1 = {}, acc2 = {}, acc3 = {};
    for (int kk = 0; kk < K; kk += 32) {
        v16h a  = load_frag(arow + kk, khalf);
        v16h b0 = load_frag(brow + kk, khalf);
        v16h b1 = load_frag(brow + (long)16 * K + kk, khalf);
        v16h b2 = load_frag(brow + (long)32 * K + kk, khalf);
        v16h b3 = load_frag(brow + (long)48 * K + kk, khalf);
        acc0 = wmma_f16(a, b0, acc0);
        acc1 = wmma_f16(a, b1, acc1);
        acc2 = wmma_f16(a, b2, acc2);
        acc3 = wmma_f16(a, b3, acc3);
    }

    // C/D layout: VGPR v -> M = v + 8*hi, N = r (within each 16x16 tile)
    int hi8 = hi << 3;
    float* crow = C + (long)nt4 * 64 + r;
#pragma unroll
    for (int v = 0; v < 8; ++v) {
        long ro = (long)(mt * 16 + hi8 + v) * Nout;
        crow[ro +  0] = acc0[v];
        crow[ro + 16] = acc1[v];
        crow[ro + 32] = acc2[v];
        crow[ro + 48] = acc3[v];
    }
}

// --------------------------------------- per-(node,head) attention logit terms
// als[n,h] = sum_c H[n, h*C+c] * a_src[h,c] ; ald analogous.
__global__ void k_att(const float* __restrict__ Hm,
                      const float* __restrict__ asr, const float* __restrict__ adr,
                      float* __restrict__ als, float* __restrict__ ald,
                      int NH, int H, int C) {
    int i = blockIdx.x * blockDim.x + threadIdx.x;
    if (i >= NH) return;
    int h = i % H;
    const float* hv = Hm + (long)i * C;   // i = n*H + h; row offset = n*H*C + h*C = i*C
    const float* av = asr + h * C;
    const float* dv = adr + h * C;
    float s = 0.f, d = 0.f;
    for (int c = 0; c < C; ++c) {
        float x = hv[c];
        s += x * av[c];
        d += x * dv[c];
    }
    als[i] = s;
    ald[i] = d;
}

// ---------------------------------------------------------------- edge passes
__device__ __forceinline__ void atomicMaxF(float* a, float v) {
    if (v >= 0.f) atomicMax((int*)a, __float_as_int(v));
    else          atomicMin((unsigned int*)a, __float_as_uint(v));
}

__device__ __forceinline__ void edge_sd(const int* src, const int* dst, int e, int E,
                                        int& s, int& d) {
    if (e < E) { s = src[e]; d = dst[e]; }
    else       { s = e - E;  d = e - E; }   // self-loops appended
}

__global__ void k_edge_max(const int* __restrict__ src, const int* __restrict__ dst,
                           int E, int Etot,
                           const float* __restrict__ als, const float* __restrict__ ald,
                           float* __restrict__ m, int H) {
    int i = blockIdx.x * blockDim.x + threadIdx.x;
    if (i >= Etot * H) return;
    int e = i / H, h = i - e * H;
    int s, d; edge_sd(src, dst, e, E, s, d);
    float ev = als[s * H + h] + ald[d * H + h];
    ev = ev > 0.f ? ev : NEG_SLOPE * ev;
    atomicMaxF(&m[d * H + h], ev);
}

__global__ void k_edge_denom(const int* __restrict__ src, const int* __restrict__ dst,
                             int E, int Etot,
                             const float* __restrict__ als, const float* __restrict__ ald,
                             const float* __restrict__ m, float* __restrict__ denom, int H) {
    int i = blockIdx.x * blockDim.x + threadIdx.x;
    if (i >= Etot * H) return;
    int e = i / H, h = i - e * H;
    int s, d; edge_sd(src, dst, e, E, s, d);
    float ev = als[s * H + h] + ald[d * H + h];
    ev = ev > 0.f ? ev : NEG_SLOPE * ev;
    atomicAdd(&denom[d * H + h], expf(ev - m[d * H + h]));
}

// one thread per (edge, 4-channel group): agg[dst] += h[src] * alpha
__global__ void k_edge_scatter(const int* __restrict__ src, const int* __restrict__ dst,
                               int E, int Etot,
                               const float* __restrict__ als, const float* __restrict__ ald,
                               const float* __restrict__ m, const float* __restrict__ denom,
                               const float* __restrict__ Hm, float* __restrict__ agg,
                               int H, int C) {
    int nout4 = (H * C) >> 2;
    int i = blockIdx.x * blockDim.x + threadIdx.x;
    if (i >= Etot * nout4) return;
    int e  = i / nout4;
    int c4 = i - e * nout4;
    int c  = c4 << 2;
    int h  = c / C;
    int s, d; edge_sd(src, dst, e, E, s, d);
    float ev = als[s * H + h] + ald[d * H + h];
    ev = ev > 0.f ? ev : NEG_SLOPE * ev;
    float alpha = expf(ev - m[d * H + h]) / (denom[d * H + h] + 1e-16f);
    const float4 hv = *(const float4*)(Hm + (long)s * (H * C) + c);
    float* ap = agg + (long)d * (H * C) + c;
    atomicAdd(ap + 0, hv.x * alpha);
    atomicAdd(ap + 1, hv.y * alpha);
    atomicAdd(ap + 2, hv.z * alpha);
    atomicAdd(ap + 3, hv.w * alpha);
}

__global__ void k_bias_act(float* __restrict__ out, const float* __restrict__ bias,
                           long n, int Nout, int doElu) {
    long i = (long)blockIdx.x * blockDim.x + threadIdx.x;
    if (i >= n) return;
    float v = out[i] + bias[i % Nout];
    if (doElu) v = v > 0.f ? v : (expf(v) - 1.f);
    out[i] = v;
}

// ---------------------------------------------------------------- host side
static inline long cdivl(long a, long b) { return (a + b - 1) / b; }

struct LayerWS {
    _Float16* Xh; _Float16* Wh;
    float *als, *ald, *m, *denom;
};

static void run_gat_layer(const float* Xin, const float* W, const float* a_s, const float* a_d,
                          const float* bias, int N, int K, int Nout, int H, int C,
                          const int* src, const int* dst, int E, int Etot,
                          float* Hbuf, float* agg, int doElu, const LayerWS& w,
                          hipStream_t stream) {
    const int TB = 256;
    // fp32 -> f16 staging
    k_f32_to_f16<<<(int)cdivl((long)N * K, TB), TB, 0, stream>>>(Xin, w.Xh, (long)N * K);
    k_f32_to_f16<<<(int)cdivl((long)Nout * K, TB), TB, 0, stream>>>(W, w.Wh, (long)Nout * K);
    // WMMA GEMM: Hbuf = X @ W^T  (each wave: 16x64 strip, 4 accumulators)
    int tilesN4 = Nout / 64, totalWaves = (N / 16) * tilesN4;
    k_gemm_wmma<<<(int)cdivl(totalWaves, 8), TB, 0, stream>>>(w.Xh, w.Wh, Hbuf, Nout, K,
                                                              tilesN4, totalWaves);
    // attention logit terms
    k_att<<<(int)cdivl((long)N * H, TB), TB, 0, stream>>>(Hbuf, a_s, a_d, w.als, w.ald,
                                                          N * H, H, C);
    // init m / denom / agg
    k_fill<<<(int)cdivl((long)N * H, TB), TB, 0, stream>>>(w.m, -INFINITY, (long)N * H);
    k_fill<<<(int)cdivl((long)N * H, TB), TB, 0, stream>>>(w.denom, 0.f, (long)N * H);
    k_fill<<<(int)cdivl((long)N * Nout, TB), TB, 0, stream>>>(agg, 0.f, (long)N * Nout);
    // segment softmax + weighted scatter
    int nEH = Etot * H;
    k_edge_max<<<(int)cdivl(nEH, TB), TB, 0, stream>>>(src, dst, E, Etot, w.als, w.ald, w.m, H);
    k_edge_denom<<<(int)cdivl(nEH, TB), TB, 0, stream>>>(src, dst, E, Etot, w.als, w.ald,
                                                         w.m, w.denom, H);
    int nSc = Etot * (Nout >> 2);
    k_edge_scatter<<<(int)cdivl(nSc, TB), TB, 0, stream>>>(src, dst, E, Etot, w.als, w.ald,
                                                           w.m, w.denom, Hbuf, agg, H, C);
    // bias (+ elu for hidden layers)
    k_bias_act<<<(int)cdivl((long)N * Nout, TB), TB, 0, stream>>>(agg, bias, (long)N * Nout,
                                                                  Nout, doElu);
}

extern "C" void kernel_launch(void* const* d_in, const int* in_sizes, int n_in,
                              void* d_out, int out_size, void* d_ws, size_t ws_size,
                              hipStream_t stream) {
    const float* x   = (const float*)d_in[0];
    const int*   ei  = (const int*)  d_in[1];
    const float* W1  = (const float*)d_in[2];
    const float* as1 = (const float*)d_in[3];
    const float* ad1 = (const float*)d_in[4];
    const float* b1  = (const float*)d_in[5];
    const float* W2  = (const float*)d_in[6];
    const float* as2 = (const float*)d_in[7];
    const float* ad2 = (const float*)d_in[8];
    const float* b2  = (const float*)d_in[9];
    const float* W3  = (const float*)d_in[10];
    const float* as3 = (const float*)d_in[11];
    const float* ad3 = (const float*)d_in[12];
    const float* b3  = (const float*)d_in[13];
    float* out = (float*)d_out;

    const int FinW = 256;                       // width of every layer input
    const int N    = in_sizes[0] / FinW;        // 50000
    const int E    = in_sizes[1] / 2;           // 800000
    const int Etot = E + N;                     // + self-loops
    const int* src = ei;
    const int* dst = ei + E;

    // workspace carve-out
    char* p = (char*)d_ws;
    float*    P   = (float*)p;    p += (size_t)N * 256 * sizeof(float);    // agg / layer output
    float*    Q   = (float*)p;    p += (size_t)N * 256 * sizeof(float);    // GEMM output h
    _Float16* Xh  = (_Float16*)p; p += (size_t)N * 256 * sizeof(_Float16);
    _Float16* Wh  = (_Float16*)p; p += (size_t)256 * 256 * sizeof(_Float16);
    float*    als = (float*)p;    p += (size_t)N * 8 * sizeof(float);
    float*    ald = (float*)p;    p += (size_t)N * 8 * sizeof(float);
    float*    m   = (float*)p;    p += (size_t)N * 8 * sizeof(float);
    float*    den = (float*)p;    p += (size_t)N * 8 * sizeof(float);
    LayerWS w{Xh, Wh, als, ald, m, den};

    // Layer 1: x[N,256] -> P[N,256]   (H=8, C=32, elu)
    run_gat_layer(x, W1, as1, ad1, b1, N, 256, 256, 8, 32,
                  src, dst, E, Etot, Q, P, /*elu=*/1, w, stream);
    // Layer 2: P -> P (via Q)         (H=8, C=32, elu)
    run_gat_layer(P, W2, as2, ad2, b2, N, 256, 256, 8, 32,
                  src, dst, E, Etot, Q, P, /*elu=*/1, w, stream);
    // Layer 3: P -> d_out[N,128]      (H=1, C=128, mean over 1 head = identity, no elu)
    run_gat_layer(P, W3, as3, ad3, b3, N, 256, 128, 1, 128,
                  src, dst, E, Etot, Q, out, /*elu=*/0, w, stream);
}
